// SelfAttention_69664369541862
// MI455X (gfx1250) — matrix operand order repaired and checked by reading, not compiled, not run on hardware
//
#include <hip/hip_runtime.h>
#include <hip/hip_bf16.h>

typedef __attribute__((ext_vector_type(16))) __bf16 v16bf;
typedef __attribute__((ext_vector_type(8)))  __bf16 v8bf;
typedef __attribute__((ext_vector_type(8)))  float  v8f;
typedef __attribute__((ext_vector_type(8)))  float  f32x8;

#define HEADS 16
#define DEPTH 64
#define EMBED 1024
#define BATCH 8
#define SEQ   1024
#define CATW  (2 * EMBED)   // 2048

// ---------------------------------------------------------------- helpers ---
__device__ __forceinline__ v8f wmma_bf16(v16bf a, v16bf b, v8f c) {
  // (neg_a, A, neg_b, B, c_mod, C, reuse_a, reuse_b)
  return __builtin_amdgcn_wmma_f32_16x16x32_bf16(false, a, false, b, (short)0, c,
                                                 false, false);
}

// load 8 contiguous f32 (32B aligned) and convert to 8 bf16
__device__ __forceinline__ v8bf cvt8(const float* p) {
  f32x8 x = *(const f32x8*)p;
  v8bf r;
#pragma unroll
  for (int i = 0; i < 8; ++i) r[i] = (__bf16)x[i];
  return r;
}

// load 8 contiguous bf16 (16B aligned)
__device__ __forceinline__ v8bf ld8(const __bf16* p) { return *(const v8bf*)p; }

__device__ __forceinline__ v16bf join8(v8bf lo, v8bf hi) {
  v16bf r;
#pragma unroll
  for (int i = 0; i < 8; ++i) { r[i] = lo[i]; r[8 + i] = hi[i]; }
  return r;
}

// reductions across a 16-lane half-group (xor masks 1,2,4,8 never cross bit 4)
__device__ __forceinline__ float redmax16(float v) {
#pragma unroll
  for (int m = 1; m < 16; m <<= 1) v = fmaxf(v, __shfl_xor(v, m, 32));
  return v;
}
__device__ __forceinline__ float redsum16(float v) {
#pragma unroll
  for (int m = 1; m < 16; m <<= 1) v += __shfl_xor(v, m, 32);
  return v;
}

// ------------------------------------------------------- weight conversion ---
// WoT[n*2048 + k] = bf16(Wo[k*1024 + n]); WxT[n*64 + k] = bf16(Wx[k*64 + n])
__global__ void __launch_bounds__(256)
cvt_weights(const float* __restrict__ Wq, const float* __restrict__ Wk,
            const float* __restrict__ Wv, const float* __restrict__ Wo,
            __bf16* __restrict__ WqT, __bf16* __restrict__ WkT,
            __bf16* __restrict__ WvT, __bf16* __restrict__ WoT) {
  const int idx = blockIdx.x * 256 + threadIdx.x;  // covers 2048*1024
  const int k = idx / EMBED;
  const int n = idx % EMBED;
  WoT[(size_t)n * CATW + k] = (__bf16)Wo[idx];
  if (idx < DEPTH * DEPTH) {
    const int kk = idx / DEPTH, nn = idx % DEPTH;
    WqT[nn * DEPTH + kk] = (__bf16)Wq[idx];
    WkT[nn * DEPTH + kk] = (__bf16)Wk[idx];
    WvT[nn * DEPTH + kk] = (__bf16)Wv[idx];
  }
}

// ------------------------------------------------------ per-head projection ---
// X: [B*L*H][64] f32 rows; WT: [n][k] bf16; one wave = 16 rows x 64 cols.
// mode 0: bf16 into cat buffer [b*SEQ+l][2048] at col h*64+n (Q)
// mode 1: bf16 row-major [B,L,H,D]                            (K)
// mode 2: bf16 transposed [B,H,D,L]                           (V)
__global__ void __launch_bounds__(256)
proj_kernel(const float* __restrict__ X, const __bf16* __restrict__ WT,
            const float* __restrict__ bias, __bf16* __restrict__ outB, int mode) {
  const int lane = threadIdx.x & 31;
  const int wave = threadIdx.x >> 5;
  const int r0   = (blockIdx.x * 8 + wave) * 16;
  const int lo   = lane & 15;
  const int hi8  = (lane < 16) ? 0 : 8;
  const int hi16 = (lane < 16) ? 0 : 16;

  const float* xrow = X + (size_t)(r0 + lo) * DEPTH;
  v16bf A[2];
#pragma unroll
  for (int c = 0; c < 2; ++c)
    A[c] = join8(cvt8(xrow + 32 * c + hi16), cvt8(xrow + 32 * c + hi16 + 8));

  const int bl = r0 >> 4;  // = b*SEQ + l (rows r0..r0+15 are the 16 heads of one (b,l))

#pragma unroll
  for (int t = 0; t < 4; ++t) {
    v8f C = {};
#pragma unroll
    for (int c = 0; c < 2; ++c) {
      const __bf16* wb = WT + (size_t)(16 * t + lo) * DEPTH + 32 * c + hi16;
      C = wmma_bf16(A[c], *(const v16bf*)wb, C);
    }
    const int n = 16 * t + lo;
    const float bv = bias[n];
    if (mode == 0) {
#pragma unroll
      for (int r = 0; r < 8; ++r) {
        const int h = r + hi8;  // row%16 == head
        outB[(size_t)bl * CATW + h * DEPTH + n] = (__bf16)(C[r] + bv);
      }
    } else if (mode == 1) {
#pragma unroll
      for (int r = 0; r < 8; ++r)
        outB[(size_t)(r0 + r + hi8) * DEPTH + n] = (__bf16)(C[r] + bv);
    } else {
      const int b = bl / SEQ, l = bl % SEQ;
#pragma unroll
      for (int r = 0; r < 8; ++r) {
        const int h = r + hi8;
        outB[((size_t)((b * HEADS + h) * DEPTH + n)) * SEQ + l] = (__bf16)(C[r] + bv);
      }
    }
  }
}

// ------------------------------------------------------- flash attention ---
// one wave per (b, h, 16-query tile); q read from cat[.][h*64+d] (bf16);
// kp bf16 [B,L,H,D]; vt bf16 [B,H,D,L]; out written bf16 into cat[.][1024+h*64+d]
__global__ void __launch_bounds__(32)
attn_kernel(const __bf16* __restrict__ kp, const __bf16* __restrict__ vt,
            __bf16* __restrict__ cat) {
  __shared__ __bf16 pbuf[16 * 32];  // P tile staging (C-layout -> A-layout)
  const int lane = threadIdx.x & 31;
  const int lo   = lane & 15;
  const int hi8  = (lane < 16) ? 0 : 8;
  const int hi16 = (lane < 16) ? 0 : 16;
  const int qt = blockIdx.x & 63;
  const int h  = (blockIdx.x >> 6) & 15;
  const int b  = blockIdx.x >> 10;
  const int q0 = qt * 16;

  const __bf16* qrow = cat + (size_t)(b * SEQ + q0 + lo) * CATW + h * DEPTH;
  v16bf aQ[2];
#pragma unroll
  for (int c = 0; c < 2; ++c)
    aQ[c] = join8(ld8(qrow + 32 * c + hi16), ld8(qrow + 32 * c + hi16 + 8));

  v8f O[4] = {};
  float mst[8], lst[8];
#pragma unroll
  for (int r = 0; r < 8; ++r) { mst[r] = -1e30f; lst[r] = 0.f; }

  const float scale = 1.0f / 32.0f;  // reference: energy / (D/2)

  for (int kt = 0; kt < SEQ / 32; ++kt) {
    const int key0 = kt * 32;
    v8f S0 = {}, S1 = {};
#pragma unroll
    for (int c = 0; c < 2; ++c) {
      const __bf16* kb0 =
          kp + ((size_t)(b * SEQ + key0 + lo) * HEADS + h) * DEPTH + 32 * c + hi16;
      S0 = wmma_bf16(aQ[c], *(const v16bf*)kb0, S0);
      const __bf16* kb1 =
          kp + ((size_t)(b * SEQ + key0 + 16 + lo) * HEADS + h) * DEPTH + 32 * c + hi16;
      S1 = wmma_bf16(aQ[c], *(const v16bf*)kb1, S1);
    }
    // online softmax; C-layout: lane<16 -> (M=r, N=lane); lane>=16 -> (M=r+8, N=lane-16)
#pragma unroll
    for (int r = 0; r < 8; ++r) {
      const float s0 = S0[r] * scale, s1 = S1[r] * scale;
      const float mx   = redmax16(fmaxf(s0, s1));
      const float mnew = fmaxf(mst[r], mx);
      const float p0 = __expf(s0 - mnew);
      const float p1 = __expf(s1 - mnew);
      const int m = r + hi8;
      pbuf[m * 32 + lo]      = (__bf16)p0;
      pbuf[m * 32 + 16 + lo] = (__bf16)p1;
      const float rs    = redsum16(p0 + p1);
      const float alpha = __expf(mst[r] - mnew);
      lst[r] = lst[r] * alpha + rs;
      mst[r] = mnew;
#pragma unroll
      for (int t = 0; t < 4; ++t) O[t][r] *= alpha;
    }
    __syncthreads();  // single-wave WG: lowers to waits (+nop barrier)
    v16bf pA = join8(*(const v8bf*)&pbuf[lo * 32 + hi16],
                     *(const v8bf*)&pbuf[lo * 32 + hi16 + 8]);
    __syncthreads();
#pragma unroll
    for (int t = 0; t < 4; ++t) {
      const __bf16* vb = vt + ((size_t)((b * HEADS + h) * DEPTH + 16 * t + lo)) * SEQ +
                         key0 + hi16;
      O[t] = wmma_bf16(pA, *(const v16bf*)vb, O[t]);
    }
  }
  // write attention output (bf16) into the second half of the cat buffer
#pragma unroll
  for (int t = 0; t < 4; ++t) {
    const int n = 16 * t + lo;
#pragma unroll
    for (int r = 0; r < 8; ++r) {
      const int m = r + hi8;
      cat[(size_t)(b * SEQ + q0 + m) * CATW + EMBED + h * DEPTH + n] =
          (__bf16)(O[t][r] / lst[r]);
    }
  }
}

// ------------------------------------------- output GEMM 8192 x 2048 x 1024 ---
// one wave = 32 rows x 64 cols (two 16-row A tiles share every B fragment)
__global__ void __launch_bounds__(256)
outproj_kernel(const __bf16* __restrict__ cat, const __bf16* __restrict__ WoT,
               const float* __restrict__ bo, float* __restrict__ out) {
  const int lane = threadIdx.x & 31;
  const int wave = threadIdx.x >> 5;
  const int w  = blockIdx.x * 8 + wave;
  const int mt = w >> 4;   // 256 dual row-tiles of 32
  const int nb = w & 15;   // 16 column blocks of 64
  const int r0 = mt * 32;
  const int n0 = nb * 64;
  const int lo   = lane & 15;
  const int hi8  = (lane < 16) ? 0 : 8;
  const int hi16 = (lane < 16) ? 0 : 16;

  v8f Ca[4] = {}, Cb[4] = {};
  const __bf16* rowA = cat + (size_t)(r0 + lo) * CATW;
  const __bf16* rowB = cat + (size_t)(r0 + 16 + lo) * CATW;

  for (int kc = 0; kc < CATW / 32; ++kc) {
    const int k0 = kc * 32;
    const v16bf A0 = join8(ld8(rowA + k0 + hi16), ld8(rowA + k0 + hi16 + 8));
    const v16bf A1 = join8(ld8(rowB + k0 + hi16), ld8(rowB + k0 + hi16 + 8));
#pragma unroll
    for (int t = 0; t < 4; ++t) {
      const v16bf Bt = *(const v16bf*)(WoT + (size_t)(n0 + 16 * t + lo) * CATW + k0 + hi16);
      Ca[t] = wmma_bf16(A0, Bt, Ca[t]);
      Cb[t] = wmma_bf16(A1, Bt, Cb[t]);
    }
  }
#pragma unroll
  for (int t = 0; t < 4; ++t) {
    const int n = n0 + 16 * t + lo;
    const float bv = bo[n];
#pragma unroll
    for (int r = 0; r < 8; ++r) {
      out[(size_t)(r0 + r + hi8) * EMBED + n]      = Ca[t][r] + bv;
      out[(size_t)(r0 + 16 + r + hi8) * EMBED + n] = Cb[t][r] + bv;
    }
  }
}

// ---------------------------------------------------------------- launcher ---
extern "C" void kernel_launch(void* const* d_in, const int* in_sizes, int n_in,
                              void* d_out, int out_size, void* d_ws, size_t ws_size,
                              hipStream_t stream) {
  const float* query = (const float*)d_in[0];
  const float* key_  = (const float*)d_in[1];
  const float* value = (const float*)d_in[2];
  // d_in[3] mask: no effect in reference (masked_fill result discarded)
  const float* Wq = (const float*)d_in[4];
  const float* bq = (const float*)d_in[5];
  const float* Wk = (const float*)d_in[6];
  const float* bk = (const float*)d_in[7];
  const float* Wv = (const float*)d_in[8];
  const float* bv = (const float*)d_in[9];
  const float* Wo = (const float*)d_in[10];
  const float* bo = (const float*)d_in[11];
  float* out = (float*)d_out;

  char* ws = (char*)d_ws;
  size_t off = 0;
  auto take = [&](size_t bytes) -> char* {
    char* p = ws + off;
    off = (off + bytes + 255) & ~(size_t)255;
    return p;
  };
  __bf16* WqT  = (__bf16*)take((size_t)DEPTH * DEPTH * 2);
  __bf16* WkT  = (__bf16*)take((size_t)DEPTH * DEPTH * 2);
  __bf16* WvT  = (__bf16*)take((size_t)DEPTH * DEPTH * 2);
  __bf16* WoT  = (__bf16*)take((size_t)CATW * EMBED * 2);             //  4 MB
  __bf16* catb = (__bf16*)take((size_t)BATCH * SEQ * CATW * 2);       // 32 MB
  __bf16* k_bf = (__bf16*)take((size_t)BATCH * SEQ * EMBED * 2);      // 16 MB
  __bf16* v_t  = (__bf16*)take((size_t)BATCH * SEQ * EMBED * 2);      // 16 MB
  (void)ws_size; (void)in_sizes; (void)n_in; (void)out_size;

  cvt_weights<<<(CATW * EMBED) / 256, 256, 0, stream>>>(Wq, Wk, Wv, Wo,
                                                        WqT, WkT, WvT, WoT);

  // B*L*H = 131072 rows -> 8192 waves -> 1024 blocks of 8 waves
  proj_kernel<<<1024, 256, 0, stream>>>(query, WqT, bq, catb, 0);
  proj_kernel<<<1024, 256, 0, stream>>>(key_,  WkT, bk, k_bf, 1);
  proj_kernel<<<1024, 256, 0, stream>>>(value, WvT, bv, v_t,  2);

  // B*H*(L/16) = 8192 single-wave workgroups
  attn_kernel<<<8192, 32, 0, stream>>>(k_bf, v_t, catb);

  // 4096 wave tiles (32x64 each) -> 512 blocks of 8 waves
  outproj_kernel<<<512, 256, 0, stream>>>(catb, WoT, bo, out);
}
